// BoxTightnessPriorLoss_63814624084548
// MI455X (gfx1250) — compile-verified, matrix operand order checked by reading, not run on hardware
//
#include <hip/hip_runtime.h>

typedef __attribute__((ext_vector_type(16))) _Float16 v16h;
typedef __attribute__((ext_vector_type(8)))  float    v8f;

#define DM      128
#define DM2     (128 * 128)
#define DM3     (128 * 128 * 128)
#define NBOX    32            // B*C*N = 2*4*4
#define NCHUNK  16            // 16 chunks of 8 w-planes
#define WPLANES 8
#define SLICEW  8
#define PER_BOX 768           // 6 * 128 float-sized slots per box
// ws layout per box (float-sized slots):
//   [0,128)   prof_d   [128,256) prof_h   [256,384) prof_w
//   [384,512) cnt_d(u32) [512,640) cnt_h(u32) [640,768) cnt_w(u32)

__global__ void k_zero(float* __restrict__ ws, float* __restrict__ out) {
    int i = blockIdx.x * blockDim.x + threadIdx.x;
    if (i < NBOX * PER_BOX) ws[i] = 0.0f;
    if (i == 0) out[0] = 0.0f;
}

// Profile along D via WMMA: column sums of masked products using an all-ones A.
// Block = (box, w-chunk). Wave wv owns d-columns [16*wv, 16*wv+16).
__global__ void k_prof_d_wmma(const float* __restrict__ logits,
                              const unsigned char* __restrict__ mask,
                              float* __restrict__ ws) {
    const int box   = blockIdx.x >> 4;
    const int chunk = blockIdx.x & 15;
    const int w0    = chunk * WPLANES;
    const int wv    = threadIdx.x >> 5;
    const int lane  = threadIdx.x & 31;
    const int half  = lane >> 4;
    const int col   = lane & 15;
    const int dBase = wv * 16;
    const int d     = dBase + col;

    const size_t logBase = (size_t)(box >> 2) * DM3;  // box/N, N=4 -> (b*C+c)*Dm^3
    const size_t mskBase = (size_t)box * DM3;

    v16h ones;
#pragma unroll
    for (int j = 0; j < 16; ++j) ones[j] = (_Float16)1.0f;

    v8f acc = {0.f, 0.f, 0.f, 0.f, 0.f, 0.f, 0.f, 0.f};
    unsigned cntLoc = 0;

    for (int it = 0; it < 32; ++it) {
        const int w  = w0 + (it >> 2);
        const int h0 = ((it & 3) << 5) + half * 16;
        const size_t rb = (size_t)w * DM2 + (size_t)h0 * DM + (size_t)d;
        if (it + 1 < 32) {  // gfx1250 global_prefetch_b8 for next row-block
            const int wn  = w0 + ((it + 1) >> 2);
            const int h0n = (((it + 1) & 3) << 5) + half * 16;
            __builtin_prefetch(&logits[logBase + (size_t)wn * DM2 + (size_t)h0n * DM + d], 0, 1);
        }
        v16h bv;
#pragma unroll
        for (int j = 0; j < 16; ++j) {
            float v = logits[logBase + rb + (size_t)j * DM];
            unsigned char m = mask[mskBase + rb + (size_t)j * DM];
            float p = m ? v : 0.0f;
            cntLoc += m;
            bv[j] = (_Float16)p;
        }
        // D = ones(16x32) * B(32x16) + C  -> every row of D = column sums of B
        acc = __builtin_amdgcn_wmma_f32_16x16x32_f16(false, ones, false, bv,
                                                     (short)0, acc, false, false);
    }

    float*    profD = ws + (size_t)box * PER_BOX;
    unsigned* cntD  = (unsigned*)(ws + (size_t)box * PER_BOX + 384);
    if (lane < 16) atomicAdd(&profD[dBase + col], acc[0]);  // D[0, col] = column sum
    atomicAdd(&cntD[d], cntLoc);
}

// Profiles along H and W. Wave <-> w-plane; lanes stride contiguous d (float4).
__global__ void k_prof_hw(const float* __restrict__ logits,
                          const unsigned char* __restrict__ mask,
                          float* __restrict__ ws) {
    __shared__ float    sProfH[DM];
    __shared__ unsigned sCntH[DM];
    const int box   = blockIdx.x >> 4;
    const int chunk = blockIdx.x & 15;
    const int wv    = threadIdx.x >> 5;
    const int lane  = threadIdx.x & 31;
    const int w     = chunk * WPLANES + wv;

    if (threadIdx.x < DM) { sProfH[threadIdx.x] = 0.f; sCntH[threadIdx.x] = 0u; }
    __syncthreads();

    const size_t logBase = (size_t)(box >> 2) * DM3 + (size_t)w * DM2;
    const size_t mskBase = (size_t)box * DM3 + (size_t)w * DM2;
    const int dOff = lane * 4;

    float wP = 0.f; unsigned wC = 0;
    for (int h = 0; h < DM; ++h) {
        const float4 v = *(const float4*)(logits + logBase + (size_t)h * DM + dOff);
        const uchar4 m = *(const uchar4*)(mask  + mskBase + (size_t)h * DM + dOff);
        float p = 0.f; unsigned c = 0;
        if (m.x) { p += v.x; ++c; }
        if (m.y) { p += v.y; ++c; }
        if (m.z) { p += v.z; ++c; }
        if (m.w) { p += v.w; ++c; }
        float rp = p; unsigned rc = c;
#pragma unroll
        for (int off = 16; off >= 1; off >>= 1) {
            rp += __shfl_xor(rp, off, 32);
            rc += __shfl_xor(rc, off, 32);
        }
        if (lane == 0) { atomicAdd(&sProfH[h], rp); atomicAdd(&sCntH[h], rc); }
        wP += p; wC += c;
    }
#pragma unroll
    for (int off = 16; off >= 1; off >>= 1) {
        wP += __shfl_xor(wP, off, 32);
        wC += __shfl_xor(wC, off, 32);
    }
    float*    profW = ws + (size_t)box * PER_BOX + 256;
    unsigned* cntW  = (unsigned*)(ws + (size_t)box * PER_BOX + 640);
    if (lane == 0) { atomicAdd(&profW[w], wP); atomicAdd(&cntW[w], wC); }

    __syncthreads();
    float*    profH = ws + (size_t)box * PER_BOX + 128;
    unsigned* cntH  = (unsigned*)(ws + (size_t)box * PER_BOX + 512);
    if (threadIdx.x < DM) {
        atomicAdd(&profH[threadIdx.x], sProfH[threadIdx.x]);
        atomicAdd(&cntH[threadIdx.x],  sCntH[threadIdx.x]);
    }
}

__global__ void k_finalize(const float* __restrict__ ws, float* __restrict__ out) {
    const int box = threadIdx.x;
    if (box >= NBOX) return;
    const float*    base  = ws + (size_t)box * PER_BOX;
    const unsigned* cbase = (const unsigned*)(base + 384);
    float e = 0.f;
#pragma unroll
    for (int a = 0; a < 3; ++a) {
        const float*    prof = base  + a * DM;
        const unsigned* cnt  = cbase + a * DM;
        for (int s = 0; s < DM / SLICEW; ++s) {
            float sv = 0.f; int sc = 0;
#pragma unroll
            for (int j = 0; j < SLICEW; ++j) {
                const int l = s * SLICEW + j;
                sv += prof[l];
                sc += (cnt[l] > 0u) ? 1 : 0;
            }
            if (sc > 0) {
                float err = 1.0f - sv / (float)sc;
                e += (err > 0.f) ? err : 0.f;
            }
        }
    }
    const float error = e * (float)SLICEW;
    atomicAdd(out, error * error);
}

extern "C" void kernel_launch(void* const* d_in, const int* in_sizes, int n_in,
                              void* d_out, int out_size, void* d_ws, size_t ws_size,
                              hipStream_t stream) {
    (void)in_sizes; (void)n_in; (void)out_size; (void)ws_size;
    const float*         logits = (const float*)d_in[0];
    const unsigned char* masks  = (const unsigned char*)d_in[1];  // jax bool -> 1 byte
    float* out = (float*)d_out;
    float* ws  = (float*)d_ws;   // needs 32*768*4 = 96 KB

    k_zero<<<96, 256, 0, stream>>>(ws, out);
    k_prof_d_wmma<<<NBOX * NCHUNK, 256, 0, stream>>>(logits, masks, ws);
    k_prof_hw<<<NBOX * NCHUNK, 256, 0, stream>>>(logits, masks, ws);
    k_finalize<<<1, 32, 0, stream>>>(ws, out);
}
